// RelativePositionMultiHeadAttention_56573309223607
// MI455X (gfx1250) — matrix operand order repaired and checked by reading, not compiled, not run on hardware
//
#include <hip/hip_runtime.h>

#define DEV __device__ __forceinline__

typedef _Float16 half_t;
typedef __attribute__((ext_vector_type(16))) _Float16 v16h;
typedef __attribute__((ext_vector_type(8)))  _Float16 v8h;
typedef __attribute__((ext_vector_type(8)))  float    v8f;

union V16H { v16h v; v8h h[2]; };

static constexpr int Bb   = 2;
static constexpr int Ss   = 2048;
static constexpr int Dd   = 1024;
static constexpr int Hh   = 16;
static constexpr int HD   = 64;
static constexpr int MTOT = Bb * Ss;      // 4096 rows
static constexpr int NBUCK = 257;         // 2*128+1
static constexpr int PROW = 40;           // padded LDS row stride (halves)

// ---- WMMA helpers -----------------------------------------------------------

DEV v8f vzero8() {
  v8f z;
  #pragma unroll
  for (int i = 0; i < 8; ++i) z[i] = 0.0f;
  return z;
}

DEV v8f wmma_f16(v16h a, v16h b, v8f c) {
  return __builtin_amdgcn_wmma_f32_16x16x32_f16(false, a, false, b,
                                                (short)0, c, false, false);
}

// A-fragment (16x32 f16): lane holds row (lane&15); K = {h8..h8+7, 16+h8..16+h8+7}
DEV v16h load_a(const half_t* base, int ld) {
  int l = threadIdx.x & 31;
  const half_t* p = base + (size_t)(l & 15) * ld + ((l >> 4) << 3);
  V16H r;
  r.h[0] = *(const v8h*)(p);
  r.h[1] = *(const v8h*)(p + 16);
  return r.v;
}

// B-fragment (32x16 f16) from Bcol[N][K] row-major: lane holds col (lane&15);
// K = {h8..h8+7, 16+h8..16+h8+7} (same order as A)
DEV v16h load_b(const half_t* base, int ld) {
  int l = threadIdx.x & 31;
  const half_t* p = base + (size_t)(l & 15) * ld + ((l >> 4) << 3);
  V16H r;
  r.h[0] = *(const v8h*)(p);
  r.h[1] = *(const v8h*)(p + 16);
  return r.v;
}

// ---- Kernel 0: fp32 -> f16 convert -----------------------------------------

__global__ void cvt_f32_f16(const float* __restrict__ src,
                            half_t* __restrict__ dst, int n) {
  int i = blockIdx.x * blockDim.x + threadIdx.x;
  int stride = gridDim.x * blockDim.x;
  for (; i < n; i += stride) dst[i] = (half_t)src[i];
}

// ---- Kernel 1: fused QKV projection ----------------------------------------
// grid (256, 3), block 256 (8 waves). Wave computes a 16x128 output tile.
// Software-pipelined: per K-chunk, prefetch next A and batch all 8 B loads
// before the 8 WMMAs so matrix ops overlap in-flight loads.

__global__ void qkv_gemm(const half_t* __restrict__ xh,
                         const half_t* __restrict__ Wqh,
                         const half_t* __restrict__ Wkh,
                         const half_t* __restrict__ Wvh,
                         const float*  __restrict__ bq,
                         const float*  __restrict__ bk,
                         const float*  __restrict__ bv,
                         half_t* __restrict__ Qh,
                         half_t* __restrict__ Kh,
                         half_t* __restrict__ Vt) {
  int mat  = blockIdx.y;
  int wave = threadIdx.x >> 5;
  int wid  = blockIdx.x * 8 + wave;     // 0..2047
  int mtile = wid >> 3;                 // 0..255
  int nr    = wid & 7;                  // 0..7 -> 128-col range
  const half_t* W    = (mat == 0) ? Wqh : (mat == 1) ? Wkh : Wvh;
  const float*  bias = (mat == 0) ? bq  : (mat == 1) ? bk  : bv;
  const half_t* Wn   = W + (size_t)nr * 128 * Dd;

  const half_t* A = xh + (size_t)mtile * 16 * Dd;
  v8f acc[8];
  #pragma unroll
  for (int t = 0; t < 8; ++t) acc[t] = vzero8();

  v16h a_cur = load_a(A, Dd);
  for (int kc = 0; kc < Dd; kc += 32) {
    int kn = (kc + 32 < Dd) ? (kc + 32) : 0;   // clamped prefetch
    v16h a_next = load_a(A + kn, Dd);
    v16h bfr[8];
    #pragma unroll
    for (int t = 0; t < 8; ++t)
      bfr[t] = load_b(Wn + (size_t)(t * 16) * Dd + kc, Dd);
    #pragma unroll
    for (int t = 0; t < 8; ++t)
      acc[t] = wmma_f16(a_cur, bfr[t], acc[t]);
    a_cur = a_next;
  }

  int l = threadIdx.x & 31;
  int nloc = l & 15, hf = l >> 4;
  #pragma unroll
  for (int t = 0; t < 8; ++t) {
    int n = nr * 128 + t * 16 + nloc;
    float bn = bias[n];
    int hh = n >> 6, hd = n & 63;
    #pragma unroll
    for (int r = 0; r < 8; ++r) {
      int m  = mtile * 16 + hf * 8 + r;
      int bb = m >> 11, ss = m & (Ss - 1);
      float val = acc[t][r] + bn;
      if (mat == 0)
        Qh[(((size_t)bb * Hh + hh) * Ss + ss) * HD + hd] = (half_t)(val * 0.125f);
      else if (mat == 1)
        Kh[(((size_t)bb * Hh + hh) * Ss + ss) * HD + hd] = (half_t)val;
      else
        Vt[(((size_t)bb * Hh + hh) * HD + hd) * Ss + ss] = (half_t)val;
    }
  }
}

// ---- Kernel 2: flash attention with relative-position bias -----------------
// grid (S/16/8 = 16, B*H = 32), block 256 (8 waves); wave owns one 16-row Q
// tile, streams keys in chunks of 32 with online softmax. K-fragments are
// double-buffered across iterations; V loads issue before the softmax VALU
// block so exp/shuffle work hides their latency.

__global__ void attn_kernel(const half_t* __restrict__ Qh,
                            const half_t* __restrict__ Kh,
                            const half_t* __restrict__ Vt,
                            const float*  __restrict__ rel_emb,
                            half_t* __restrict__ attn) {
  __shared__ float  biascol[NBUCK];
  __shared__ half_t plds[8][16 * PROW];

  int bh = blockIdx.y;               // 0..31
  int b  = bh >> 4, h = bh & 15;
  for (int i = threadIdx.x; i < NBUCK; i += blockDim.x)
    biascol[i] = rel_emb[i * Hh + h];
  __syncthreads();

  int wave = threadIdx.x >> 5;
  int l    = threadIdx.x & 31;
  int nloc = l & 15, hf = l >> 4;
  int q0   = (blockIdx.x * 8 + wave) * 16;

  const half_t* Qb = Qh + (size_t)bh * Ss * HD;
  const half_t* Kb = Kh + (size_t)bh * Ss * HD;
  const half_t* Vb = Vt + (size_t)bh * HD * Ss;

  v16h qf0 = load_a(Qb + (size_t)q0 * HD, HD);       // dims 0..31
  v16h qf1 = load_a(Qb + (size_t)q0 * HD + 32, HD);  // dims 32..63

  v8f o[4];
  float mstat[8], lstat[8];
  #pragma unroll
  for (int r = 0; r < 8; ++r) { mstat[r] = -1e30f; lstat[r] = 0.0f; }
  #pragma unroll
  for (int nb = 0; nb < 4; ++nb) o[nb] = vzero8();

  half_t* pw = &plds[wave][0];

  // preload K fragments for chunk 0
  v16h kf0 = load_b(Kb, HD);
  v16h kf1 = load_b(Kb + 32, HD);
  v16h kf2 = load_b(Kb + (size_t)16 * HD, HD);
  v16h kf3 = load_b(Kb + (size_t)16 * HD + 32, HD);

  for (int k0 = 0; k0 < Ss; k0 += 32) {
    // ---- scores (scale already folded into Q) ----
    v8f c0 = vzero8(), c1 = vzero8();
    c0 = wmma_f16(qf0, kf0, c0);
    c0 = wmma_f16(qf1, kf1, c0);
    c1 = wmma_f16(qf0, kf2, c1);
    c1 = wmma_f16(qf1, kf3, c1);

    // ---- prefetch V fragments for this chunk (consumed after softmax) ----
    v16h vf0 = load_b(Vb + k0, Ss);
    v16h vf1 = load_b(Vb + (size_t)16 * Ss + k0, Ss);
    v16h vf2 = load_b(Vb + (size_t)32 * Ss + k0, Ss);
    v16h vf3 = load_b(Vb + (size_t)48 * Ss + k0, Ss);

    // ---- prefetch K fragments for next chunk (clamped) ----
    int kn = (k0 + 32 < Ss) ? (k0 + 32) : 0;
    kf0 = load_b(Kb + (size_t)kn * HD, HD);
    kf1 = load_b(Kb + (size_t)kn * HD + 32, HD);
    kf2 = load_b(Kb + (size_t)(kn + 16) * HD, HD);
    kf3 = load_b(Kb + (size_t)(kn + 16) * HD + 32, HD);

    // ---- relative-position bias (uniform fast path off the +-128 band) ----
    int dlo = q0 - (k0 + 31);
    int dhi = q0 + 15 - k0;
    if (dlo >= 128) {
      float bc = biascol[256];
      #pragma unroll
      for (int r = 0; r < 8; ++r) { c0[r] += bc; c1[r] += bc; }
    } else if (dhi <= -128) {
      float bc = biascol[0];
      #pragma unroll
      for (int r = 0; r < 8; ++r) { c0[r] += bc; c1[r] += bc; }
    } else {
      int kg0 = k0 + nloc, kg1 = kg0 + 16;
      #pragma unroll
      for (int r = 0; r < 8; ++r) {
        int qg = q0 + hf * 8 + r;
        int d0 = qg - kg0; d0 = d0 < -128 ? -128 : (d0 > 128 ? 128 : d0);
        int d1 = qg - kg1; d1 = d1 < -128 ? -128 : (d1 > 128 ? 128 : d1);
        c0[r] += biascol[d0 + 128];
        c1[r] += biascol[d1 + 128];
      }
    }

    // ---- online softmax (row reductions within 16-lane halves) ----
    #pragma unroll
    for (int r = 0; r < 8; ++r) {
      float v = fmaxf(c0[r], c1[r]);
      v = fmaxf(v, __shfl_xor(v, 1, 32));
      v = fmaxf(v, __shfl_xor(v, 2, 32));
      v = fmaxf(v, __shfl_xor(v, 4, 32));
      v = fmaxf(v, __shfl_xor(v, 8, 32));
      float mnew  = fmaxf(mstat[r], v);
      float alpha = __expf(mstat[r] - mnew);
      mstat[r] = mnew;
      float p0 = __expf(c0[r] - mnew);
      float p1 = __expf(c1[r] - mnew);
      float srow = p0 + p1;
      srow += __shfl_xor(srow, 1, 32);
      srow += __shfl_xor(srow, 2, 32);
      srow += __shfl_xor(srow, 4, 32);
      srow += __shfl_xor(srow, 8, 32);
      lstat[r] = lstat[r] * alpha + srow;
      #pragma unroll
      for (int nb = 0; nb < 4; ++nb) o[nb][r] *= alpha;
      c0[r] = p0; c1[r] = p1;
    }

    // ---- P: D-layout -> A-fragment via per-wave LDS staging ----
    #pragma unroll
    for (int r = 0; r < 8; ++r) {
      int m = hf * 8 + r;
      pw[m * PROW + nloc]      = (half_t)c0[r];
      pw[m * PROW + 16 + nloc] = (half_t)c1[r];
    }
    __asm__ volatile("s_wait_dscnt 0" ::: "memory");
    v16h pf;
    {
      const half_t* p = pw + (l & 15) * PROW + ((l >> 4) << 3);
      V16H t;
      t.h[0] = *(const v8h*)(p);
      t.h[1] = *(const v8h*)(p + 16);
      pf = t.v;
    }

    // ---- O += P @ V ----
    o[0] = wmma_f16(pf, vf0, o[0]);
    o[1] = wmma_f16(pf, vf1, o[1]);
    o[2] = wmma_f16(pf, vf2, o[2]);
    o[3] = wmma_f16(pf, vf3, o[3]);
  }

  // ---- finalize: divide by row sum, store f16 in [B*S, D] for out-proj ----
  #pragma unroll
  for (int r = 0; r < 8; ++r) {
    float inv = 1.0f / lstat[r];
    size_t row = (size_t)b * Ss + (size_t)(q0 + hf * 8 + r);
    #pragma unroll
    for (int nb = 0; nb < 4; ++nb)
      attn[row * Dd + h * HD + nb * 16 + nloc] = (half_t)(o[nb][r] * inv);
  }
}

// ---- Kernel 3: output projection -> fp32 d_out -----------------------------

__global__ void out_gemm(const half_t* __restrict__ ah,
                         const half_t* __restrict__ Woh,
                         const float*  __restrict__ bo,
                         float* __restrict__ out) {
  int wave = threadIdx.x >> 5;
  int wid  = blockIdx.x * 8 + wave;
  int mtile = wid >> 3, nr = wid & 7;
  const half_t* A  = ah + (size_t)mtile * 16 * Dd;
  const half_t* Wn = Woh + (size_t)nr * 128 * Dd;
  v8f acc[8];
  #pragma unroll
  for (int t = 0; t < 8; ++t) acc[t] = vzero8();

  v16h a_cur = load_a(A, Dd);
  for (int kc = 0; kc < Dd; kc += 32) {
    int kn = (kc + 32 < Dd) ? (kc + 32) : 0;
    v16h a_next = load_a(A + kn, Dd);
    v16h bfr[8];
    #pragma unroll
    for (int t = 0; t < 8; ++t)
      bfr[t] = load_b(Wn + (size_t)(t * 16) * Dd + kc, Dd);
    #pragma unroll
    for (int t = 0; t < 8; ++t)
      acc[t] = wmma_f16(a_cur, bfr[t], acc[t]);
    a_cur = a_next;
  }

  int l = threadIdx.x & 31;
  int nloc = l & 15, hf = l >> 4;
  #pragma unroll
  for (int t = 0; t < 8; ++t) {
    int n = nr * 128 + t * 16 + nloc;
    float bn = bo[n];
    #pragma unroll
    for (int r = 0; r < 8; ++r) {
      int m = mtile * 16 + hf * 8 + r;
      out[(size_t)m * Dd + n] = acc[t][r] + bn;
    }
  }
}

// ---- Host launcher ----------------------------------------------------------

extern "C" void kernel_launch(void* const* d_in, const int* in_sizes, int n_in,
                              void* d_out, int out_size, void* d_ws, size_t ws_size,
                              hipStream_t stream) {
  (void)in_sizes; (void)n_in; (void)out_size; (void)ws_size;

  const float* x   = (const float*)d_in[0];
  const float* Wq  = (const float*)d_in[1];
  const float* bq  = (const float*)d_in[2];
  const float* Wk  = (const float*)d_in[3];
  const float* bk  = (const float*)d_in[4];
  const float* Wv  = (const float*)d_in[5];
  const float* bv  = (const float*)d_in[6];
  const float* Wo  = (const float*)d_in[7];
  const float* bo  = (const float*)d_in[8];
  const float* rel = (const float*)d_in[9];
  float* out = (float*)d_out;

  char* ws = (char*)d_ws;
  half_t* xh  = (half_t*)(ws);                       // 8 MiB  [4096,1024]
  half_t* Wqh = (half_t*)(ws + ( 8ull << 20));       // 2 MiB
  half_t* Wkh = (half_t*)(ws + (10ull << 20));       // 2 MiB
  half_t* Wvh = (half_t*)(ws + (12ull << 20));       // 2 MiB
  half_t* Woh = (half_t*)(ws + (14ull << 20));       // 2 MiB
  half_t* Qh  = (half_t*)(ws + (16ull << 20));       // 8 MiB  [B,H,S,64]
  half_t* Kh  = (half_t*)(ws + (24ull << 20));       // 8 MiB  [B,H,S,64]
  half_t* Vt  = (half_t*)(ws + (32ull << 20));       // 8 MiB  [B,H,64,S]
  half_t* ah  = (half_t*)(ws + (40ull << 20));       // 8 MiB  [4096,1024]

  // 0) converts
  cvt_f32_f16<<<1024, 256, 0, stream>>>(x,  xh,  MTOT * Dd);
  cvt_f32_f16<<<512,  256, 0, stream>>>(Wq, Wqh, Dd * Dd);
  cvt_f32_f16<<<512,  256, 0, stream>>>(Wk, Wkh, Dd * Dd);
  cvt_f32_f16<<<512,  256, 0, stream>>>(Wv, Wvh, Dd * Dd);
  cvt_f32_f16<<<512,  256, 0, stream>>>(Wo, Woh, Dd * Dd);

  // 1) fused QKV projection
  qkv_gemm<<<dim3(256, 3), 256, 0, stream>>>(xh, Wqh, Wkh, Wvh,
                                             bq, bk, bv, Qh, Kh, Vt);

  // 2) flash attention with relative position bias
  attn_kernel<<<dim3(16, 32), 256, 0, stream>>>(Qh, Kh, Vt, rel, ah);

  // 3) output projection
  out_gemm<<<dim3(256), 256, 0, stream>>>(ah, Woh, bo, out);
}
